// TokenizeBlock_75050258530767
// MI455X (gfx1250) — matrix operand order repaired and checked by reading, not compiled, not run on hardware
//
#include <hip/hip_runtime.h>
#include <math.h>

typedef __attribute__((ext_vector_type(16))) __bf16 bf16x16;
typedef __attribute__((ext_vector_type(8)))  __bf16 bf16x8;
typedef __attribute__((ext_vector_type(8)))  float  f32x8;

#define BM        64      // C rows per block
#define BK        32      // K per WMMA step (bf16)
#define NOUT      256     // output features
#define EFEAT     256     // input features per token
#define KTOT      1024    // 4 * EFEAT
#define NIT       (KTOT / BK)
#define ASTR      40      // LDS row stride in halves (32 + 8 pad, keeps 16B align)
#define BSTR      40

// ---------------------------------------------------------------------------
// Prep kernel: W fp32 [K][N] -> Wb bf16 [N][K]  (512 KB, written once per call)
// Thread t of block b handles column n=t, rows k0..k0+63; reads are coalesced
// across threads per row, writes are 128B contiguous per thread.
// ---------------------------------------------------------------------------
__global__ __launch_bounds__(256)
void convert_W_bf16(const float* __restrict__ W, unsigned short* __restrict__ wb_raw)
{
    __bf16* Wb = (__bf16*)wb_raw;
    const int n  = threadIdx.x;
    const int k0 = blockIdx.x * 64;
    bf16x8 hv[8];
    #pragma unroll
    for (int r = 0; r < 64; ++r)
        hv[r >> 3][r & 7] = (__bf16)W[(long)(k0 + r) * NOUT + n];
    bf16x8* dst = (bf16x8*)(Wb + (long)n * KTOT + k0);
    #pragma unroll
    for (int i = 0; i < 8; ++i) dst[i] = hv[i];
}

// ---------------------------------------------------------------------------
// Fused LayerNorm -> token-merge -> bf16 WMMA GEMM -> bias + exact GELU
// ---------------------------------------------------------------------------
template <bool USE_WB>
__global__ __launch_bounds__(256)
void fused_ln_tokmerge_dense_gelu(const float* __restrict__ x,
                                  const float* __restrict__ gamma,
                                  const float* __restrict__ beta,
                                  const float* __restrict__ W,
                                  const float* __restrict__ bias,
                                  const unsigned short* __restrict__ wb_raw,
                                  float* __restrict__ out)
{
    __shared__ alignas(16) __bf16 As[2][BM * ASTR];     // 2 x 5.1 KB  [row][k]
    __shared__ alignas(16) __bf16 Bs[2][NOUT * BSTR];   // 2 x 20.5 KB [n][k]
    __shared__ float s_mean[BM * 4];
    __shared__ float s_rs[BM * 4];
    __shared__ float s_gamma[EFEAT];
    __shared__ float s_beta[EFEAT];

    const __bf16* Wb = (const __bf16*)wb_raw;

    const int tid  = threadIdx.x;
    const int lane = tid & 31;
    const int wave = tid >> 5;          // 0..7
    const int M0   = blockIdx.x * BM;   // first C row of this block

    s_gamma[tid] = gamma[tid];
    s_beta[tid]  = beta[tid];

    // ---------------- Phase 0: LayerNorm stats for 256 tokens ----------------
    for (int i = 0; i < 32; ++i) {
        const int tl = wave * 32 + i;                       // local token 0..255
        const long gt = (long)(M0 * 4 + tl);
        const float4* p = (const float4*)(x + gt * EFEAT + lane * 8);
        float4 v0 = p[0], v1 = p[1];
        float s = v0.x + v0.y + v0.z + v0.w + v1.x + v1.y + v1.z + v1.w;
        float q = v0.x*v0.x + v0.y*v0.y + v0.z*v0.z + v0.w*v0.w
                + v1.x*v1.x + v1.y*v1.y + v1.z*v1.z + v1.w*v1.w;
        #pragma unroll
        for (int off = 16; off > 0; off >>= 1) {
            s += __shfl_xor(s, off, 32);
            q += __shfl_xor(q, off, 32);
        }
        if (lane == 0) {
            float mean = s * (1.0f / 256.0f);
            float var  = q * (1.0f / 256.0f) - mean * mean;
            s_mean[tl] = mean;
            s_rs[tl]   = rsqrtf(var + 1e-6f);
        }
    }

    // ---------------- Phase 1: double-buffered bf16 WMMA GEMM ----------------
    const int wm   = wave >> 2;         // 0..1 -> 32-row sub-tile
    const int wn   = wave & 3;          // 0..3 -> 64-col sub-tile
    const int half = lane >> 4;
    const int l15  = lane & 15;
    const int kbA  = half * 8;          // A frag: halves {kbA..kbA+7, kbA+16..kbA+23}
    const int kbB  = half * 8;          // B frag: halves {kbB..kbB+7, kbB+16..kbB+23}

    const int arow = tid >> 2;          // 0..63 : A loader row
    const int acb  = (tid & 3) * 8;     // 0,8,16,24 : A loader col base

    f32x8 acc[2][4];
    #pragma unroll
    for (int mi = 0; mi < 2; ++mi)
        #pragma unroll
        for (int ni = 0; ni < 4; ++ni)
            acc[mi][ni] = (f32x8)(0.0f);

    // ---- helpers inlined twice (prologue + steady state) via lambdas ----
    auto stageA = [&](int it, int buf) {
        const int k0 = it * BK, j = k0 >> 8, fb = k0 & 255;
        const int tl = arow * 4 + j;
        const float* src = x + (long)((M0 + arow) * 4 + j) * EFEAT + fb + acb;
        float4 v0 = ((const float4*)src)[0];
        float4 v1 = ((const float4*)src)[1];
        const float mean = s_mean[tl], rs = s_rs[tl];
        const float* g  = &s_gamma[fb + acb];
        const float* bt = &s_beta[fb + acb];
        float t[8] = {v0.x, v0.y, v0.z, v0.w, v1.x, v1.y, v1.z, v1.w};
        bf16x8 hv;
        #pragma unroll
        for (int e = 0; e < 8; ++e)
            hv[e] = (__bf16)(((t[e] - mean) * rs) * g[e] + bt[e]);
        *(bf16x8*)(&As[buf][arow * ASTR + acb]) = hv;
    };
    auto stageB = [&](int it, int buf) {
        const int k0 = it * BK;
        if (USE_WB) {
            const bf16x8* wp = (const bf16x8*)(Wb + (long)tid * KTOT + k0);
            bf16x8 b0 = wp[0], b1 = wp[1], b2 = wp[2], b3 = wp[3];
            bf16x8* dst = (bf16x8*)(&Bs[buf][tid * BSTR]);
            dst[0] = b0; dst[1] = b1;
            *(bf16x8*)(&Bs[buf][tid * BSTR + 16]) = b2;
            *(bf16x8*)(&Bs[buf][tid * BSTR + 24]) = b3;
            if (it + 1 < NIT)
                __builtin_prefetch((const void*)(Wb + (long)tid * KTOT + k0 + BK), 0, 1);
        } else {
            const float* wp = W + (long)k0 * NOUT + tid;
            #pragma unroll
            for (int kk = 0; kk < BK; kk += 8) {
                bf16x8 hv;
                #pragma unroll
                for (int e = 0; e < 8; ++e)
                    hv[e] = (__bf16)wp[(kk + e) * NOUT];
                *(bf16x8*)(&Bs[buf][tid * BSTR + kk]) = hv;
            }
        }
    };

    __syncthreads();            // stats visible
    stageA(0, 0);
    stageB(0, 0);

    int buf = 0;
    for (int it = 0; it < NIT; ++it) {
        __syncthreads();        // stage(it)->buf visible; buf^1 free to overwrite

        // fragments for this chunk
        bf16x16 afrag[2], bfrag[4];
        #pragma unroll
        for (int mi = 0; mi < 2; ++mi) {
            const __bf16* base = &As[buf][(wm * 32 + mi * 16 + l15) * ASTR + kbA];
            bf16x8 lo = *(const bf16x8*)(base);
            bf16x8 hi = *(const bf16x8*)(base + 16);
            afrag[mi] = __builtin_shufflevector(lo, hi,
                0,1,2,3,4,5,6,7,8,9,10,11,12,13,14,15);
        }
        #pragma unroll
        for (int ni = 0; ni < 4; ++ni) {
            const __bf16* base = &Bs[buf][(wn * 64 + ni * 16 + l15) * BSTR + kbB];
            bf16x8 lo = *(const bf16x8*)(base);
            bf16x8 hi = *(const bf16x8*)(base + 16);
            bfrag[ni] = __builtin_shufflevector(lo, hi,
                0,1,2,3,4,5,6,7,8,9,10,11,12,13,14,15);
        }

        // stage next chunk into the other buffer (overlaps with WMMAs below)
        if (it + 1 < NIT) {
            stageA(it + 1, buf ^ 1);
            stageB(it + 1, buf ^ 1);
        }

        #pragma unroll
        for (int mi = 0; mi < 2; ++mi)
            #pragma unroll
            for (int ni = 0; ni < 4; ++ni)
                acc[mi][ni] = __builtin_amdgcn_wmma_f32_16x16x32_bf16(
                    false, afrag[mi], false, bfrag[ni],
                    (short)0, acc[mi][ni], false, false);

        buf ^= 1;
    }

    // ---------------- Epilogue: bias + exact GELU, fp32 stores ----------------
    #pragma unroll
    for (int mi = 0; mi < 2; ++mi) {
        #pragma unroll
        for (int ni = 0; ni < 4; ++ni) {
            const int col = wn * 64 + ni * 16 + l15;
            const float bv = bias[col];
            #pragma unroll
            for (int v = 0; v < 8; ++v) {
                const int row = M0 + wm * 32 + mi * 16 + v + half * 8;
                float val = acc[mi][ni][v] + bv;
                val = 0.5f * val * (1.0f + erff(val * 0.70710678118654752f));
                out[(long)row * NOUT + col] = val;
            }
        }
    }
}

extern "C" void kernel_launch(void* const* d_in, const int* in_sizes, int n_in,
                              void* d_out, int out_size, void* d_ws, size_t ws_size,
                              hipStream_t stream) {
    const float* x     = (const float*)d_in[0];
    const float* gamma = (const float*)d_in[1];
    const float* beta  = (const float*)d_in[2];
    const float* W     = (const float*)d_in[3];
    const float* b     = (const float*)d_in[4];
    float* out = (float*)d_out;

    const int tokens = in_sizes[0] / EFEAT;   // B*S = 262144
    const int M = tokens / 4;                 // 65536 output rows
    dim3 grid(M / BM), block(256);

    const size_t wbBytes = (size_t)NOUT * KTOT * sizeof(unsigned short); // 512 KB
    if (ws_size >= wbBytes) {
        unsigned short* Wb = (unsigned short*)d_ws;
        convert_W_bf16<<<dim3(KTOT / 64), dim3(256), 0, stream>>>(W, Wb);
        fused_ln_tokmerge_dense_gelu<true><<<grid, block, 0, stream>>>(
            x, gamma, beta, W, b, Wb, out);
    } else {
        fused_ln_tokmerge_dense_gelu<false><<<grid, block, 0, stream>>>(
            x, gamma, beta, W, b, (const unsigned short*)nullptr, out);
    }
}
